// Neural_Network_lstmcell_51634096832691
// MI455X (gfx1250) — matrix operand order repaired and checked by reading, not compiled, not run on hardware
//
#include <hip/hip_runtime.h>
#include <math.h>

// ---------------------------------------------------------------------------
// Types
// ---------------------------------------------------------------------------
typedef __attribute__((ext_vector_type(16))) _Float16 v16h;
typedef __attribute__((ext_vector_type(8)))  _Float16 v8h;
typedef __attribute__((ext_vector_type(8)))  float    v8f;
typedef __attribute__((ext_vector_type(4)))  unsigned int v4u;
typedef __attribute__((ext_vector_type(8)))  int v8i;
typedef __attribute__((ext_vector_type(4)))  int v4i;

#define CAT16(lo, hi) __builtin_shufflevector((lo), (hi), 0,1,2,3,4,5,6,7,8,9,10,11,12,13,14,15)

// Problem constants
#define TT   256
#define BB   2048
#define DIN  136      // 32 mlp_tmpl + 16 mlp_static + 88 emb
#define H1   128
#define H2   32
#define H3   16

// Combined weight shapes (f16)
#define K1   264      // DIN + H1
#define K2   160      // H1 + H2

// ws layout (bytes)
#define ENCH_BYTES  142606336ull            // 524288 * 136 * 2
#define WC_BYTES    311296ull               // 512*264*2 + 128*160*2
#define BC1_OFF     (ENCH_BYTES + WC_BYTES)
#define BC2_OFF     (BC1_OFF + 2048ull)

// LDS layout (bytes)
#define WC1_OFF   0          // 512*264*2 = 270336
#define WC2_OFF   270336     // 128*160*2 = 40960
#define XB_OFF    311296     // [2][16][136] f16 = 8704
#define H1_OFF    320000     // [16][152]  f16 = 4864   (128 real + 24 zero pad)
#define H2_OFF    324864     // [16][32]   f16 = 1024
#define SMEM_BYTES 325888

#if __has_builtin(__builtin_amdgcn_tensor_load_to_lds) && __has_builtin(__builtin_amdgcn_s_wait_tensorcnt)
#define USE_TDM 1
#else
#define USE_TDM 0
#endif

// Fast activations: one v_exp_f32 + one v_rcp_f32 each, saturate correctly.
__device__ __forceinline__ float sigmoidf_(float x) {
    return __builtin_amdgcn_rcpf(1.f + __expf(-x));
}
__device__ __forceinline__ float tanhf_(float x) {
    // tanh(x) = 1 - 2/(exp(2x)+1); exp->inf => 1, exp->0 => -1 (no NaN)
    return 1.f - 2.f * __builtin_amdgcn_rcpf(__expf(2.f * x) + 1.f);
}

// ---------------------------------------------------------------------------
// Prep kernel 1: pack enc = [mlp_tmpl(32) | mlp_static(16) | emb(88)] as f16.
// Flat row r == (t*B + b) == (i*256 + j) under the reference's .view trick.
// ---------------------------------------------------------------------------
__global__ void prep_ench(const float* __restrict__ mlp_s, const float* __restrict__ mlp_t,
                          const int* __restrict__ emb_s, const int* __restrict__ emb_t,
                          const float* __restrict__ t0, const float* __restrict__ t1,
                          const float* __restrict__ t2, const float* __restrict__ t3,
                          const float* __restrict__ t4, const float* __restrict__ t5,
                          _Float16* __restrict__ ench)
{
    size_t idx = (size_t)blockIdx.x * blockDim.x + threadIdx.x;
    if (idx >= (size_t)TT * BB * DIN) return;
    size_t r = idx / DIN;
    int    c = (int)(idx % DIN);
    float  v;
    if (c < 32)      v = mlp_t[r * 32 + c];
    else if (c < 48) v = mlp_s[r * 16 + (c - 32)];
    else {
        int tbl, base, width;
        if (c < 80)       { tbl = 0; base = 48;  width = 32; }
        else if (c < 96)  { tbl = 1; base = 80;  width = 16; }
        else if (c < 112) { tbl = 2; base = 96;  width = 16; }
        else if (c < 120) { tbl = 3; base = 112; width = 8;  }
        else if (c < 128) { tbl = 4; base = 120; width = 8;  }
        else              { tbl = 5; base = 128; width = 8;  }
        int col = c - base;
        int ix  = (tbl < 4) ? emb_t[r * 4 + tbl] : emb_s[r * 2 + (tbl - 4)];
        const float* tp = (tbl == 0 ? t0 : tbl == 1 ? t1 : tbl == 2 ? t2 : tbl == 3 ? t3 : tbl == 4 ? t4 : t5);
        v = tp[(size_t)ix * width + col];
    }
    ench[idx] = (_Float16)v;
}

// ---------------------------------------------------------------------------
// Prep kernel 2: combined f16 weights [W_ih|W_hh] and fused f32 biases.
// ---------------------------------------------------------------------------
__global__ void prep_w(const float* __restrict__ Wih1, const float* __restrict__ Whh1,
                       const float* __restrict__ Wih2, const float* __restrict__ Whh2,
                       const float* __restrict__ bih1, const float* __restrict__ bhh1,
                       const float* __restrict__ bih2, const float* __restrict__ bhh2,
                       _Float16* __restrict__ wc, float* __restrict__ bc1, float* __restrict__ bc2)
{
    int idx = blockIdx.x * blockDim.x + threadIdx.x;
    if (idx < 512 * K1) {
        int r = idx / K1, c = idx % K1;
        float v = (c < DIN) ? Wih1[r * DIN + c] : Whh1[r * H1 + (c - DIN)];
        wc[idx] = (_Float16)v;
    } else if (idx < 512 * K1 + 128 * K2) {
        int k = idx - 512 * K1;
        int r = k / K2, c = k % K2;
        float v = (c < H1) ? Wih2[r * H1 + c] : Whh2[r * H2 + (c - H1)];
        wc[idx] = (_Float16)v;
    } else if (idx < 512 * K1 + 128 * K2 + 512) {
        int k = idx - (512 * K1 + 128 * K2);
        bc1[k] = bih1[k] + bhh1[k];
    } else if (idx < 512 * K1 + 128 * K2 + 512 + 128) {
        int k = idx - (512 * K1 + 128 * K2 + 512);
        bc2[k] = bih2[k] + bhh2[k];
    }
}

// ---------------------------------------------------------------------------
// Main kernel: one block = 16 sequences, 8 waves, 256-step recurrence.
// LSTM1: wave w owns gate rows [16w,16w+16) of all four gates (tiles w,w+8,w+16,w+24).
// LSTM2: waves 0/1 own rows [16v,16v+16) of all four gates (tiles 2g+v).
// Final 16x32 linear + dot: wave 2.
// ---------------------------------------------------------------------------
__global__ __launch_bounds__(256)
void lstm_main(const _Float16* __restrict__ ench,
               const _Float16* __restrict__ wcg,
               const float* __restrict__ bc1g, const float* __restrict__ bc2g,
               const float* __restrict__ W1,  const float* __restrict__ b1,
               const float* __restrict__ W2,  const float* __restrict__ b2,
               float* __restrict__ out)
{
    extern __shared__ char smem[];
    const int tid  = threadIdx.x;
    const int wv   = tid >> 5;
    const int lane = tid & 31;
    const int n    = lane & 15;   // column (sequence within tile) / A-row
    const int hl   = lane >> 4;   // lane half
    const int i0   = blockIdx.x * 16;

    char* wc1 = smem + WC1_OFF;
    char* wc2 = smem + WC2_OFF;
    char* h1b = smem + H1_OFF;
    char* h2b = smem + H2_OFF;

    // --- cooperative: weights -> LDS, zero h buffers, load x(0) ---
    {
        const v4u* src = (const v4u*)wcg;
        v4u* dst = (v4u*)smem;
        for (int k = tid; k < (int)(WC_BYTES / 16); k += 256) dst[k] = src[k];
        v4u z = {0u, 0u, 0u, 0u};
        v4u* zb = (v4u*)h1b;                       // h1 (4864) + h2 (1024) contiguous
        for (int k = tid; k < (4864 + 1024) / 16; k += 256) zb[k] = z;
        const v4u* xs = (const v4u*)ench;          // 17 x uint4 per 136-half row
        v4u* xd = (v4u*)(smem + XB_OFF);
        for (int k = tid; k < 272; k += 256) {
            int row = k / 17, p = k % 17;
            xd[k] = xs[((size_t)(i0 + row) * TT + 0) * 17 + p];
        }
    }
    __syncthreads();

    // --- per-wave constant fragments (biases, final-layer weights) ---
    v8f bf1[4], bf2[4];
#pragma unroll
    for (int g = 0; g < 4; ++g)
#pragma unroll
        for (int k = 0; k < 8; ++k) {
            bf1[g][k] = bc1g[g * H1 + wv * 16 + hl * 8 + k];
            bf2[g][k] = (wv < 2) ? bc2g[g * H2 + wv * 16 + hl * 8 + k] : 0.f;
        }
    v16h aW1;
#pragma unroll
    for (int k = 0; k < 8; ++k) {
        aW1[k]     = (_Float16)W1[n * 32 + hl * 16 + k];
        aW1[8 + k] = (_Float16)W1[n * 32 + hl * 16 + 8 + k];
    }
    v8f bf3; float w2v[8];
#pragma unroll
    for (int k = 0; k < 8; ++k) { bf3[k] = b1[hl * 8 + k]; w2v[k] = W2[hl * 8 + k]; }
    const float b2s = b2[0];

    v8f c1 = {0.f,0.f,0.f,0.f,0.f,0.f,0.f,0.f};
    v8f c2 = {0.f,0.f,0.f,0.f,0.f,0.f,0.f,0.f};

    for (int j = 0; j < TT; ++j) {
        const int buf = j & 1;
        char* xb  = smem + XB_OFF + buf * 4352;
        char* xbn = smem + XB_OFF + (1 - buf) * 4352;

        // ---- async prefetch of x(j+1) into the other buffer ----
        if (j < TT - 1) {
#if USE_TDM
            if (wv == 0) {
                unsigned long long ga = (unsigned long long)(uintptr_t)ench +
                    (unsigned long long)((size_t)i0 * TT + (j + 1)) * (DIN * 2ull);
                v4u g0;
                g0[0] = 1u;                                        // count=1
                g0[1] = (unsigned)(uintptr_t)xbn;                  // lds_addr
                g0[2] = (unsigned)ga;                              // global_addr lo
                g0[3] = ((unsigned)(ga >> 32) & 0x1FFFFFFu) | (2u << 30); // hi + type=2
                v8i g1;
                g1[0] = (int)(1u << 16);          // data_size = 2B
                g1[1] = (int)(136u << 16);        // tensor_dim0 lo16
                g1[2] = (int)(16u << 16);         // tensor_dim1 lo16
                g1[3] = (int)(136u << 16);        // tile_dim0
                g1[4] = 16;                       // tile_dim1
                g1[5] = TT * DIN;                 // tensor_dim0_stride = 34816 elements
                g1[6] = 0; g1[7] = 0;
                v4i z4 = {0, 0, 0, 0};
#if __clang_major__ >= 23
                v8i z8 = {0, 0, 0, 0, 0, 0, 0, 0};
                __builtin_amdgcn_tensor_load_to_lds(g0, g1, z4, z4, z8, 0);
#else
                __builtin_amdgcn_tensor_load_to_lds(g0, g1, z4, z4, 0);
#endif
            }
#else
            {
                const v4u* xs = (const v4u*)ench;
                v4u* xd = (v4u*)xbn;
                for (int k = tid; k < 272; k += 256) {
                    int row = k / 17, p = k % 17;
                    xd[k] = xs[((size_t)(i0 + row) * TT + (j + 1)) * 17 + p];
                }
            }
#endif
        }

        // ---- LSTM1: 4 gate tiles x 9 K-chunks of 16x16x32 f16 WMMA ----
        v8f d0 = bf1[0], d1 = bf1[1], d2g = bf1[2], d3 = bf1[3];
#pragma unroll
        for (int c = 0; c < 9; ++c) {
            int f0 = c * 32 + hl * 16;
            const char* p0 = (f0 < DIN)
                ? (xb  + ((unsigned)n * 136 + f0) * 2)
                : (h1b + ((unsigned)n * 152 + (f0 - DIN)) * 2);
            const char* p1 = (f0 + 8 < DIN)
                ? (xb  + ((unsigned)n * 136 + f0 + 8) * 2)
                : (h1b + ((unsigned)n * 152 + (f0 + 8 - DIN)) * 2);
            v16h bfr = CAT16(*(const v8h*)p0, *(const v8h*)p1);
#pragma unroll
            for (int g = 0; g < 4; ++g) {
                int row = g * H1 + wv * 16 + n;
                v8h lo, hi;
                if (c < 8) {
                    lo = *(const v8h*)(wc1 + ((unsigned)row * K1 + c * 32 + hl * 16) * 2);
                    hi = *(const v8h*)(wc1 + ((unsigned)row * K1 + c * 32 + hl * 16 + 8) * 2);
                } else {
                    // ragged last chunk: only cols 256..263 exist
                    v8h t = *(const v8h*)(wc1 + ((unsigned)row * K1 + 256) * 2);
                    v8h z = {(_Float16)0,(_Float16)0,(_Float16)0,(_Float16)0,
                             (_Float16)0,(_Float16)0,(_Float16)0,(_Float16)0};
                    lo = hl ? z : t;
                    hi = z;
                }
                v16h afr = CAT16(lo, hi);
                v8f& d = (g == 0 ? d0 : g == 1 ? d1 : g == 2 ? d2g : d3);
                d = __builtin_amdgcn_wmma_f32_16x16x32_f16(false, afr, false, bfr,
                                                           (short)0, d, false, false);
            }
        }
        // gate elementwise: c1' = sig(f)*c1 + sig(i)*tanh(g); h1 = sig(o)*tanh(c1')
        v8h hpack;
#pragma unroll
        for (int k = 0; k < 8; ++k) {
            float ii = sigmoidf_(d0[k]);
            float ff = sigmoidf_(d1[k]);
            float gg = tanhf_(d2g[k]);
            float oo = sigmoidf_(d3[k]);
            float cc = ff * c1[k] + ii * gg;
            c1[k] = cc;
            hpack[k] = (_Float16)(oo * tanhf_(cc));
        }
        __syncthreads();                                          // readers of old h1 done
        *(v8h*)(h1b + ((unsigned)n * 152 + wv * 16 + hl * 8) * 2) = hpack;
        __syncthreads();                                          // new h1 visible

        // ---- LSTM2 (waves 0,1): 4 gate tiles x 5 K-chunks ----
        v8h h2pack;
        if (wv < 2) {
            v8f e0 = bf2[0], e1 = bf2[1], e2 = bf2[2], e3 = bf2[3];
#pragma unroll
            for (int c = 0; c < 5; ++c) {
                int f0 = c * 32 + hl * 16;
                const char* q0 = (f0 < H1)
                    ? (h1b + ((unsigned)n * 152 + f0) * 2)
                    : (h2b + ((unsigned)n * 32 + (f0 - H1)) * 2);
                const char* q1 = (f0 + 8 < H1)
                    ? (h1b + ((unsigned)n * 152 + f0 + 8) * 2)
                    : (h2b + ((unsigned)n * 32 + (f0 + 8 - H1)) * 2);
                v16h bfr = CAT16(*(const v8h*)q0, *(const v8h*)q1);
#pragma unroll
                for (int g = 0; g < 4; ++g) {
                    int row = (2 * g + wv) * 16 + n;
                    v8h lo = *(const v8h*)(wc2 + ((unsigned)row * K2 + c * 32 + hl * 16) * 2);
                    v8h hi = *(const v8h*)(wc2 + ((unsigned)row * K2 + c * 32 + hl * 16 + 8) * 2);
                    v16h afr = CAT16(lo, hi);
                    v8f& e = (g == 0 ? e0 : g == 1 ? e1 : g == 2 ? e2 : e3);
                    e = __builtin_amdgcn_wmma_f32_16x16x32_f16(false, afr, false, bfr,
                                                               (short)0, e, false, false);
                }
            }
#pragma unroll
            for (int k = 0; k < 8; ++k) {
                float ii = sigmoidf_(e0[k]);
                float ff = sigmoidf_(e1[k]);
                float gg = tanhf_(e2[k]);
                float oo = sigmoidf_(e3[k]);
                float cc = ff * c2[k] + ii * gg;
                c2[k] = cc;
                h2pack[k] = (_Float16)(oo * tanhf_(cc));
            }
        }
        __syncthreads();                                          // readers of old h2 done
        if (wv < 2)
            *(v8h*)(h2b + ((unsigned)n * 32 + wv * 16 + hl * 8) * 2) = h2pack;
        __syncthreads();                                          // new h2 visible

        // ---- head (wave 2): relu(W1 @ h2 + b1) -> sigmoid(W2 @ . + b2) ----
        if (wv == 2) {
            const char* q = h2b + ((unsigned)n * 32 + hl * 16) * 2;
            v16h bfr = CAT16(*(const v8h*)q, *(const v8h*)(q + 16));
            v8f lin = __builtin_amdgcn_wmma_f32_16x16x32_f16(false, aW1, false, bfr,
                                                             (short)0, bf3, false, false);
            float acc = 0.f;
#pragma unroll
            for (int k = 0; k < 8; ++k) acc += w2v[k] * fmaxf(lin[k], 0.f);
            acc += __shfl_xor(acc, 16, 32);
            if (hl == 0) out[(size_t)(i0 + n) * TT + j] = sigmoidf_(acc + b2s);
        }

#if USE_TDM
        if (wv == 0) __builtin_amdgcn_s_wait_tensorcnt(0);        // x(j+1) landed
#endif
        __syncthreads();                                          // step boundary
    }
}

// ---------------------------------------------------------------------------
extern "C" void kernel_launch(void* const* d_in, const int* in_sizes, int n_in,
                              void* d_out, int out_size, void* d_ws, size_t ws_size,
                              hipStream_t stream)
{
    (void)in_sizes; (void)n_in; (void)out_size; (void)ws_size;
    const float* mlp_s = (const float*)d_in[0];
    const float* mlp_t = (const float*)d_in[1];
    const int*   emb_s = (const int*)d_in[2];
    const int*   emb_t = (const int*)d_in[3];
    const float* t0 = (const float*)d_in[4];
    const float* t1 = (const float*)d_in[5];
    const float* t2 = (const float*)d_in[6];
    const float* t3 = (const float*)d_in[7];
    const float* t4 = (const float*)d_in[8];
    const float* t5 = (const float*)d_in[9];
    const float* Wih1 = (const float*)d_in[10];
    const float* bih1 = (const float*)d_in[11];
    const float* Whh1 = (const float*)d_in[12];
    const float* bhh1 = (const float*)d_in[13];
    const float* Wih2 = (const float*)d_in[14];
    const float* bih2 = (const float*)d_in[15];
    const float* Whh2 = (const float*)d_in[16];
    const float* bhh2 = (const float*)d_in[17];
    const float* W1 = (const float*)d_in[18];
    const float* b1 = (const float*)d_in[19];
    const float* W2 = (const float*)d_in[20];
    const float* b2 = (const float*)d_in[21];

    char* ws = (char*)d_ws;
    _Float16* ench = (_Float16*)ws;
    _Float16* wc   = (_Float16*)(ws + ENCH_BYTES);
    float*    bc1  = (float*)(ws + BC1_OFF);
    float*    bc2  = (float*)(ws + BC2_OFF);

    prep_ench<<<278528, 256, 0, stream>>>(mlp_s, mlp_t, emb_s, emb_t,
                                          t0, t1, t2, t3, t4, t5, ench);
    prep_w<<<611, 256, 0, stream>>>(Wih1, Whh1, Wih2, Whh2,
                                    bih1, bhh1, bih2, bhh2, wc, bc1, bc2);
    lstm_main<<<128, 256, SMEM_BYTES, stream>>>(ench, wc, bc1, bc2,
                                                W1, b1, W2, b2, (float*)d_out);
}
